// AydinMoETensoric_455266534075
// MI455X (gfx1250) — matrix operand order repaired and checked by reading, not compiled, not run on hardware
//
#include <hip/hip_runtime.h>
#include <hip/hip_bf16.h>

// ---------------- problem constants ----------------
constexpr int E_ = 8;
constexpr int H_ = 1024;
constexpr int I_ = 2048;
constexpr int T_ = 8192;        // B*S tokens
constexpr int KTOP = 2;

// ---------------- WMMA types ----------------
typedef __attribute__((ext_vector_type(16))) __bf16       v16bf;
typedef __attribute__((ext_vector_type(8)))  float        v8f;
typedef __attribute__((ext_vector_type(8)))  unsigned int v8u;

static __device__ inline v8f wmma_bf16(const v8u& a, const v8u& b, v8f c) {
  return __builtin_amdgcn_wmma_f32_16x16x32_bf16(
      false, __builtin_bit_cast(v16bf, a),
      false, __builtin_bit_cast(v16bf, b),
      (short)0, c, false, false);
}

static __device__ inline unsigned int bf16u(float f) {
  unsigned int u = __float_as_uint(f);
  return ((u + 0x7FFFu + ((u >> 16) & 1u)) >> 16) & 0xFFFFu; // RNE fp32->bf16
}
static __device__ inline unsigned int pack2bf(float lo, float hi) {
  return bf16u(lo) | (bf16u(hi) << 16);
}

// ---------------- CDNA5 async copy global->LDS (ASYNCcnt tracked) ----------------
#define USE_ASYNC 1
static __device__ inline void async_b128(void* lds_dst, const void* gsrc) {
#if USE_ASYNC
  unsigned int       lds = (unsigned int)(unsigned long long)(uintptr_t)lds_dst;
  unsigned long long ga  = (unsigned long long)(uintptr_t)gsrc;
  asm volatile("global_load_async_to_lds_b128 %0, %1, off"
               :: "v"(lds), "v"(ga) : "memory");
#else
  *(uint4*)lds_dst = *(const uint4*)gsrc;
#endif
}
static __device__ inline void async_wait() {
#if USE_ASYNC
  asm volatile("s_wait_asynccnt 0" ::: "memory");
#endif
}

// ---------------- 0: zero counters ----------------
__global__ void moe_init(int* counts, int* cursors) {
  int t = threadIdx.x;
  if (t < E_) { counts[t] = 0; cursors[t] = 0; }
}

// ---------------- 1: router (one wave32 per token) ----------------
__global__ __launch_bounds__(256) void moe_router(
    const float* __restrict__ x, const float* __restrict__ rw,
    int* __restrict__ topk_idx, float* __restrict__ topk_w, int* counts) {
  int wave = (blockIdx.x * blockDim.x + threadIdx.x) >> 5;
  int lane = threadIdx.x & 31;
  if (wave >= T_) return;
  const float* xr = x + (size_t)wave * H_;
  float acc[E_] = {0.f,0.f,0.f,0.f,0.f,0.f,0.f,0.f};
  for (int h = lane; h < H_; h += 32) {
    float xv = xr[h];
#pragma unroll
    for (int e = 0; e < E_; ++e) acc[e] += xv * rw[e * H_ + h];
  }
#pragma unroll
  for (int e = 0; e < E_; ++e) {
#pragma unroll
    for (int off = 16; off > 0; off >>= 1) acc[e] += __shfl_xor(acc[e], off, 32);
  }
  if (lane == 0) {
    float mx = acc[0];
#pragma unroll
    for (int e = 1; e < E_; ++e) mx = fmaxf(mx, acc[e]);
    float p[E_]; float den = 0.f;
#pragma unroll
    for (int e = 0; e < E_; ++e) { p[e] = __expf(acc[e] - mx); den += p[e]; }
    int i0 = 0;
#pragma unroll
    for (int e = 1; e < E_; ++e) if (p[e] > p[i0]) i0 = e;
    int i1 = (i0 == 0) ? 1 : 0;
#pragma unroll
    for (int e = 0; e < E_; ++e) if (e != i0 && p[e] > p[i1]) i1 = e;
    float w0 = p[i0] / den, w1 = p[i1] / den;
    float s = w0 + w1 + 1e-6f;
    topk_idx[2 * wave + 0] = i0;  topk_idx[2 * wave + 1] = i1;
    topk_w[2 * wave + 0] = w0 / s; topk_w[2 * wave + 1] = w1 / s;
    atomicAdd(&counts[i0], 1); atomicAdd(&counts[i1], 1);
  }
}

// ---------------- 2: prefix sum over 8 experts ----------------
__global__ void moe_prefix(const int* counts, int* base) {
  if (threadIdx.x == 0) {
    int s = 0;
    for (int e = 0; e < E_; ++e) { base[e] = s; s += counts[e]; }
    base[E_] = s;
  }
}

// ---------------- 3: scatter tokens into expert buckets ----------------
__global__ void moe_scatter(const int* __restrict__ topk_idx, int* cursors,
                            const int* __restrict__ base,
                            int* __restrict__ token_list, int* __restrict__ row_of) {
  int t = blockIdx.x * blockDim.x + threadIdx.x;
  if (t >= T_) return;
#pragma unroll
  for (int k = 0; k < KTOP; ++k) {
    int e = topk_idx[2 * t + k];
    int pos = atomicAdd(&cursors[e], 1);
    int row = base[e] + pos;
    token_list[row] = t;
    row_of[2 * t + k] = row;
  }
}

// ---------------- 4a: convert x -> packed bf16 pairs [T][H/2] ----------------
__global__ __launch_bounds__(256) void conv_x(const float* __restrict__ x,
                                              unsigned int* __restrict__ xb) {
  int gid = blockIdx.x * 256 + threadIdx.x;        // T_*H_/2 total
  float2 v = ((const float2*)x)[gid];
  xb[gid] = pack2bf(v.x, v.y);
}

// ---------------- 4b: transpose+convert weights: [E][K][N] f32 -> [E][N][K/2] uint(bf16x2)
__global__ __launch_bounds__(256) void conv_wT(const float* __restrict__ w,
                                               unsigned int* __restrict__ wT,
                                               int K, int N) {
  int e = blockIdx.z;
  const float* we = w + (size_t)e * K * N;
  unsigned int* wTe = wT + (size_t)e * N * (K >> 1);
  int k0 = blockIdx.y * 64, n0 = blockIdx.x * 64;
  __shared__ float tile[64][65];
  for (int i = threadIdx.x; i < 64 * 64; i += 256) {
    int kk = i >> 6, nn = i & 63;
    tile[kk][nn] = we[(size_t)(k0 + kk) * N + n0 + nn];
  }
  __syncthreads();
  for (int i = threadIdx.x; i < 64 * 32; i += 256) {
    int nn = i >> 5, kp = i & 31;
    wTe[(size_t)(n0 + nn) * (K >> 1) + (k0 >> 1) + kp] =
        pack2bf(tile[2 * kp][nn], tile[2 * kp + 1][nn]);
  }
}

// ---------------- GEMM tiling ----------------
#define BM 128             // block rows
#define BN 64              // block cols
#define BK 32              // K per stage
#define KP (BK / 2)        // 16 packed-uint pairs per tile row
#define LDP 20             // padded LDS row stride in uints (80B: 16B aligned, conflict-free)

// fragment builders per CDNA5 ISA §7.12.2 layouts; `sub` = 16-row subtile index
static __device__ inline v8u frag_a(const unsigned int (*As)[LDP], int sub, int lane) {
  int m = sub * 16 + (lane & 15);
  int ak = (lane >> 4) * 4;
  v8u a;
#pragma unroll
  for (int i = 0; i < 8; ++i) {
    int kpi = ((i < 4) ? 0 : 8) + ak + (i & 3);
    a[i] = As[m][kpi];
  }
  return a;
}
static __device__ inline v8u frag_b(const unsigned int (*Bs)[LDP], int nbase, int lane) {
  int n = nbase + (lane & 15);
  int bk = (lane >> 4) * 4;
  v8u b;
#pragma unroll
  for (int i = 0; i < 8; ++i) b[i] = Bs[n][((i < 4) ? 0 : 8) + bk + (i & 3)];
  return b;
}

// ---------------- 5: gemm1 — gathered X(bf16) @ w13T[e], fused SwiGLU -> h_buf (bf16)
// 8 waves: wm = wv>>1 in 0..3 (32 rows each), wn = wv&1 (32 cols each) => 8 WMMA/wave/K-step
__global__ __launch_bounds__(256) void moe_gemm1(
    const unsigned int* __restrict__ xb,     // [T][H/2]
    const unsigned int* __restrict__ w13T,   // [E][2I][H/2]
    const int* __restrict__ base, const int* __restrict__ counts,
    const int* __restrict__ token_list, unsigned short* __restrict__ h_buf) {
  int e  = blockIdx.z;
  int nE = counts[e];
  int m0 = blockIdx.y * BM;
  if (m0 >= nE) return;
  int n0 = blockIdx.x * BN;                     // column within [0, I)
  int eb = base[e];
  const int* toks = token_list + eb;
  const unsigned int* we = w13T + (size_t)e * (2 * I_) * (H_ / 2);

  __shared__ __align__(16) unsigned int As[2][BM][LDP];
  __shared__ __align__(16) unsigned int Bg[2][BN][LDP];
  __shared__ __align__(16) unsigned int Bu[2][BN][LDP];

  int tid = threadIdx.x, wv = tid >> 5, lane = tid & 31;
  int wm = wv >> 1, wn = wv & 1;

  // stage: one b128 per thread per 64x16-uint tile (A needs two: 128 rows)
  int snl = tid >> 2, sq = (tid & 3) * 4;
  int ar0 = m0 + snl;      if (ar0 >= nE) ar0 = nE - 1;   // clamped gather (stores guarded)
  int ar1 = m0 + snl + 64; if (ar1 >= nE) ar1 = nE - 1;
  const unsigned int* asrc0 = xb + (size_t)toks[ar0] * (H_ / 2) + sq;
  const unsigned int* asrc1 = xb + (size_t)toks[ar1] * (H_ / 2) + sq;
  const unsigned int* gsrc  = we + (size_t)(n0 + snl) * (H_ / 2) + sq;
  const unsigned int* usrc  = we + (size_t)(I_ + n0 + snl) * (H_ / 2) + sq;

  auto stage = [&](int buf, int k0) {
    int kq = k0 >> 1;
    async_b128(&As[buf][snl][sq],      asrc0 + kq);
    async_b128(&As[buf][snl + 64][sq], asrc1 + kq);
    async_b128(&Bg[buf][snl][sq],      gsrc + kq);
    async_b128(&Bu[buf][snl][sq],      usrc + kq);
  };

  v8f cg[2][2] = {{{}, {}}, {{}, {}}};
  v8f cu[2][2] = {{{}, {}}, {{}, {}}};
  stage(0, 0);
  int buf = 0;
  for (int k0 = 0; k0 < H_; k0 += BK) {
    async_wait();
    __syncthreads();
    if (k0 + BK < H_) stage(buf ^ 1, k0 + BK);

    v8u a0 = frag_a(As[buf], 2 * wm + 0, lane);
    v8u a1 = frag_a(As[buf], 2 * wm + 1, lane);
    v8u g0 = frag_b(Bg[buf], wn * 32,      lane);
    v8u g1 = frag_b(Bg[buf], wn * 32 + 16, lane);
    v8u u0 = frag_b(Bu[buf], wn * 32,      lane);
    v8u u1 = frag_b(Bu[buf], wn * 32 + 16, lane);
    cg[0][0] = wmma_bf16(a0, g0, cg[0][0]);
    cg[0][1] = wmma_bf16(a0, g1, cg[0][1]);
    cg[1][0] = wmma_bf16(a1, g0, cg[1][0]);
    cg[1][1] = wmma_bf16(a1, g1, cg[1][1]);
    cu[0][0] = wmma_bf16(a0, u0, cu[0][0]);
    cu[0][1] = wmma_bf16(a0, u1, cu[0][1]);
    cu[1][0] = wmma_bf16(a1, u0, cu[1][0]);
    cu[1][1] = wmma_bf16(a1, u1, cu[1][1]);
    buf ^= 1;
  }

  // epilogue: h = silu(gate) * up -> bf16 h_buf[(eb+row)*I + n]
#pragma unroll
  for (int mi = 0; mi < 2; ++mi) {
    int mrow0 = m0 + (2 * wm + mi) * 16 + (lane >> 4) * 8;
    int ncol0 = n0 + wn * 32 + (lane & 15);
#pragma unroll
    for (int t = 0; t < 2; ++t) {
#pragma unroll
      for (int r = 0; r < 8; ++r) {
        int row = mrow0 + r;
        if (row < nE) {
          float g = cg[mi][t][r], u = cu[mi][t][r];
          float hv = g * __builtin_amdgcn_rcpf(1.f + __expf(-g)) * u;
          h_buf[(size_t)(eb + row) * I_ + (ncol0 + t * 16)] =
              (unsigned short)bf16u(hv);
        }
      }
    }
  }
}

// ---------------- 6: gemm2 — h_buf(bf16) @ w2T[e] -> y_buf (fp32) ----------------
__global__ __launch_bounds__(256) void moe_gemm2(
    const unsigned int* __restrict__ hu,     // h_buf as packed pairs [ROWS][I/2]
    const unsigned int* __restrict__ w2T,    // [E][H][I/2]
    const int* __restrict__ base, const int* __restrict__ counts,
    float* __restrict__ y_buf) {
  int e  = blockIdx.z;
  int nE = counts[e];
  int m0 = blockIdx.y * BM;
  if (m0 >= nE) return;
  int n0 = blockIdx.x * BN;                    // column within [0, H)
  int eb = base[e];
  const unsigned int* we = w2T + (size_t)e * H_ * (I_ / 2);

  __shared__ __align__(16) unsigned int As[2][BM][LDP];
  __shared__ __align__(16) unsigned int Bs[2][BN][LDP];

  int tid = threadIdx.x, wv = tid >> 5, lane = tid & 31;
  int wm = wv >> 1, wn = wv & 1;

  int snl = tid >> 2, sq = (tid & 3) * 4;
  int ar0 = m0 + snl;      if (ar0 >= nE) ar0 = nE - 1;
  int ar1 = m0 + snl + 64; if (ar1 >= nE) ar1 = nE - 1;
  const unsigned int* asrc0 = hu + (size_t)(eb + ar0) * (I_ / 2) + sq;
  const unsigned int* asrc1 = hu + (size_t)(eb + ar1) * (I_ / 2) + sq;
  const unsigned int* bsrc  = we + (size_t)(n0 + snl) * (I_ / 2) + sq;

  auto stage = [&](int buf, int k0) {
    int kq = k0 >> 1;
    async_b128(&As[buf][snl][sq],      asrc0 + kq);
    async_b128(&As[buf][snl + 64][sq], asrc1 + kq);
    async_b128(&Bs[buf][snl][sq],      bsrc + kq);
  };

  v8f c[2][2] = {{{}, {}}, {{}, {}}};
  stage(0, 0);
  int buf = 0;
  for (int k0 = 0; k0 < I_; k0 += BK) {
    async_wait();
    __syncthreads();
    if (k0 + BK < I_) stage(buf ^ 1, k0 + BK);

    v8u a0 = frag_a(As[buf], 2 * wm + 0, lane);
    v8u a1 = frag_a(As[buf], 2 * wm + 1, lane);
    v8u b0 = frag_b(Bs[buf], wn * 32,      lane);
    v8u b1 = frag_b(Bs[buf], wn * 32 + 16, lane);
    c[0][0] = wmma_bf16(a0, b0, c[0][0]);
    c[0][1] = wmma_bf16(a0, b1, c[0][1]);
    c[1][0] = wmma_bf16(a1, b0, c[1][0]);
    c[1][1] = wmma_bf16(a1, b1, c[1][1]);
    buf ^= 1;
  }

#pragma unroll
  for (int mi = 0; mi < 2; ++mi) {
    int mrow0 = m0 + (2 * wm + mi) * 16 + (lane >> 4) * 8;
    int ncol0 = n0 + wn * 32 + (lane & 15);
#pragma unroll
    for (int t = 0; t < 2; ++t) {
#pragma unroll
      for (int r = 0; r < 8; ++r) {
        int row = mrow0 + r;
        if (row < nE)
          y_buf[(size_t)(eb + row) * H_ + (ncol0 + t * 16)] = c[mi][t][r];
      }
    }
  }
}

// ---------------- 7: combine top-2 expert rows ----------------
__global__ __launch_bounds__(256) void moe_combine(
    const float* __restrict__ y_buf, const int* __restrict__ row_of,
    const float* __restrict__ topk_w, float* __restrict__ out) {
  int gid = blockIdx.x * 256 + threadIdx.x;   // T_*H_ total
  int t = gid >> 10;                          // H_ = 1024
  int h = gid & 1023;
  int r0 = row_of[2 * t], r1 = row_of[2 * t + 1];
  float w0 = topk_w[2 * t], w1 = topk_w[2 * t + 1];
  out[gid] = w0 * y_buf[(size_t)r0 * H_ + h] + w1 * y_buf[(size_t)r1 * H_ + h];
}

// ---------------- host ----------------
extern "C" void kernel_launch(void* const* d_in, const int* in_sizes, int n_in,
                              void* d_out, int out_size, void* d_ws, size_t ws_size,
                              hipStream_t stream) {
  const float* x   = (const float*)d_in[0];   // (B,S,H)
  const float* rw  = (const float*)d_in[1];   // (E,H)
  const float* w13 = (const float*)d_in[2];   // (E,H,2I)
  const float* w2  = (const float*)d_in[3];   // (E,I,H)
  float* out = (float*)d_out;

  char* ws = (char*)d_ws;
  int*   topk_idx   = (int*)(ws);                        // 64KB
  float* topk_w     = (float*)(ws + (64 << 10));         // 64KB
  int*   counts     = (int*)(ws + (128 << 10));
  int*   cursors    = counts + 8;
  int*   ebase      = cursors + 8;
  int*   token_list = (int*)(ws + (132 << 10));          // 64KB
  int*   row_of     = (int*)(ws + (196 << 10));          // 64KB
  unsigned short* h_buf = (unsigned short*)(ws + (512 << 10));             // 64MB
  float*          y_buf = (float*)(ws + (512 << 10) + (64ull << 20));      // 64MB
  unsigned int*   xb    = (unsigned int*)(ws + (512 << 10) + (128ull << 20)); // 16MB
  unsigned int*   w13T  = (unsigned int*)(ws + (512 << 10) + (144ull << 20)); // 64MB
  unsigned int*   w2T   = (unsigned int*)(ws + (512 << 10) + (208ull << 20)); // 32MB

  moe_init   <<<1, 32, 0, stream>>>(counts, cursors);
  moe_router <<<T_ / 8, 256, 0, stream>>>(x, rw, topk_idx, topk_w, counts);
  moe_prefix <<<1, 1, 0, stream>>>(counts, ebase);
  moe_scatter<<<T_ / 256, 256, 0, stream>>>(topk_idx, cursors, ebase,
                                            token_list, row_of);
  conv_x     <<<(T_ * H_ / 2) / 256, 256, 0, stream>>>(x, xb);
  conv_wT    <<<dim3((2 * I_) / 64, H_ / 64, E_), 256, 0, stream>>>(w13, w13T, H_, 2 * I_);
  conv_wT    <<<dim3(H_ / 64, I_ / 64, E_), 256, 0, stream>>>(w2, w2T, I_, H_);
  moe_gemm1  <<<dim3(I_ / BN, T_ / BM, E_), 256, 0, stream>>>(
      xb, w13T, ebase, counts, token_list, h_buf);
  moe_gemm2  <<<dim3(H_ / BN, T_ / BM, E_), 256, 0, stream>>>(
      (const unsigned int*)h_buf, w2T, ebase, counts, y_buf);
  moe_combine<<<(T_ * H_) / 256, 256, 0, stream>>>(y_buf, row_of, topk_w, out);
  (void)in_sizes; (void)n_in; (void)out_size; (void)ws_size;
}
